// CoRoPE_90426241450288
// MI455X (gfx1250) — matrix operand order repaired and checked by reading, not compiled, not run on hardware
//
#include <hip/hip_runtime.h>
#include <hip/hip_bf16.h>
#include <math.h>

// ---------------------------------------------------------------------------
// GQA attention block for MI455X (gfx1250, wave32, WMMA).
// Pipeline: fp32->bf16 convert, Q/K/V GEMMs (bf16 WMMA), flash attention
// (bf16 WMMA + fp32 online softmax), output GEMM (fp32 out).
// Tile staging uses gfx1250 GLOBAL_LOAD_ASYNC_TO_LDS (ASYNCcnt) when the
// toolchain exposes the builtins, else falls back to register round-trip.
// ---------------------------------------------------------------------------

typedef __attribute__((ext_vector_type(16))) __bf16 v16bf;
typedef __attribute__((ext_vector_type(8)))  __bf16 v8bf;
typedef __attribute__((ext_vector_type(4)))  __bf16 v4bf;
typedef __attribute__((ext_vector_type(8)))  float  v8f;
typedef __attribute__((ext_vector_type(4)))  int    v4i;

#define WMMA_BF16(a, b, c) \
  __builtin_amdgcn_wmma_f32_16x16x32_bf16(false, (a), false, (b), (short)0, (c), false, false)

#if defined(__gfx1250__) && __has_builtin(__builtin_amdgcn_global_load_async_to_lds_b128) && \
    __has_builtin(__builtin_amdgcn_s_wait_asynccnt)
#define USE_ASYNC_LDS 1
#else
#define USE_ASYNC_LDS 0
#endif

#if USE_ASYNC_LDS
typedef __attribute__((address_space(1))) v4i* as1_v4i_p;  // global int4*
typedef __attribute__((address_space(3))) v4i* as3_v4i_p;  // LDS int4*
#endif

// Copy 16 bytes global -> LDS. Async path issues the DMA-style async load
// (tracked with ASYNCcnt); fallback is a VGPR round-trip.
__device__ __forceinline__ void stage16(const __bf16* g, __bf16* l) {
#if USE_ASYNC_LDS
  __builtin_amdgcn_global_load_async_to_lds_b128((as1_v4i_p)g, (as3_v4i_p)l, 0, 0);
#else
  *(uint4*)l = *(const uint4*)g;
#endif
}

__device__ __forceinline__ void stage_wait() {
#if USE_ASYNC_LDS
  __builtin_amdgcn_s_wait_asynccnt(0);
#endif
}

constexpr int BATCH = 2;
constexpr int SEQ   = 2048;
constexpr int EMB   = 2048;
constexpr int HD    = 128;          // head dim
constexpr int KVE   = 512;          // kv embed = 4 * 128
constexpr int NH    = 16;
constexpr int MTOT  = BATCH * SEQ;  // 4096 rows
constexpr float SCALE = 0.08838834764831845f;  // 1/sqrt(128)

// ---------------------------------------------------------------------------
// Fragment loaders from LDS tiles (layouts per CDNA5 ISA §7.12.2).
// A (16x32, 16-bit): lane holds row M=lane&15; K chunks [kb,kb+8) and
//                    [kb+16,kb+24) with kb = (lane>>4)*8.
// B (32x16, 16-bit): lane holds col N=lane&15; K chunks [kh,kh+8) and
//                    [kh+16,kh+24) with kh = (lane>>4)*8.
// ---------------------------------------------------------------------------
__device__ __forceinline__ v16bf load_a_frag(const __bf16* tile, int stride, int lane) {
  const int r  = lane & 15;
  const int kb = (lane >> 4) << 3;
  const __bf16* p = tile + r * stride + kb;
  v8bf lo = *(const v8bf*)(p);
  v8bf hi = *(const v8bf*)(p + 16);
  v16bf f;
#pragma unroll
  for (int i = 0; i < 8; ++i) { f[i] = lo[i]; f[i + 8] = hi[i]; }
  return f;
}

__device__ __forceinline__ v16bf load_b_frag(const __bf16* tile, int stride, int lane) {
  const int n  = lane & 15;
  const int kh = (lane >> 4) << 3;
  const __bf16* p = tile + n * stride + kh;
  v8bf lo = *(const v8bf*)(p);
  v8bf hi = *(const v8bf*)(p + 16);
  v16bf f;
#pragma unroll
  for (int i = 0; i < 8; ++i) { f[i] = lo[i]; f[i + 8] = hi[i]; }
  return f;
}

// ---------------------------------------------------------------------------
// fp32 -> bf16 conversion (vectorized x4)
// ---------------------------------------------------------------------------
__global__ void cvt_f32_bf16(const float* __restrict__ in, __bf16* __restrict__ out, int n4) {
  const int i = blockIdx.x * blockDim.x + threadIdx.x;
  if (i < n4) {
    const float4 f = ((const float4*)in)[i];
    v4bf o;
    o[0] = (__bf16)f.x; o[1] = (__bf16)f.y; o[2] = (__bf16)f.z; o[3] = (__bf16)f.w;
    ((v4bf*)out)[i] = o;
  }
}

// ---------------------------------------------------------------------------
// GEMM: Out[m][n] = sum_k A[m][k] * W[n][k]   (i.e. A @ W^T)
// A: MxK row-major bf16, W: NxK row-major bf16 (torch Linear weight layout).
// Block tile 128x64, BK=32. 8 waves; each wave owns a 32x32 accumulation
// (2x2 WMMA tiles). OUT_F32 selects fp32 vs bf16 output.
// ---------------------------------------------------------------------------
template <bool OUT_F32>
__global__ __launch_bounds__(256) void gemm_bf16(const __bf16* __restrict__ A,
                                                 const __bf16* __restrict__ W,
                                                 void* __restrict__ Out,
                                                 int M, int N, int K) {
  constexpr int BM = 128, BN = 64, BK = 32, PAD = 8;
  __shared__ __bf16 sA[BM][BK + PAD];
  __shared__ __bf16 sB[BN][BK + PAD];

  const int tid  = threadIdx.x;
  const int lane = tid & 31;
  const int wave = tid >> 5;      // 0..7
  const int wm   = wave & 3;      // 4 waves along M
  const int wn   = wave >> 2;     // 2 waves along N
  const int m0   = blockIdx.y * BM;
  const int n0   = blockIdx.x * BN;

  const v8f vzero = {0.f, 0.f, 0.f, 0.f, 0.f, 0.f, 0.f, 0.f};
  v8f acc[2][2];
#pragma unroll
  for (int i = 0; i < 2; ++i)
#pragma unroll
    for (int j = 0; j < 2; ++j) acc[i][j] = vzero;

  // Cooperative tile load mapping: A 128x32 (16 elems/thread), B 64x32 (8/thread)
  const int arow = tid >> 1, acol = (tid & 1) << 4;
  const int brow = tid >> 2, bcol = (tid & 3) << 3;
  const __bf16* ga = A + (size_t)(m0 + arow) * K + acol;
  const __bf16* gb = W + (size_t)(n0 + brow) * K + bcol;

  for (int k0 = 0; k0 < K; k0 += BK) {
    stage16(ga + k0,     &sA[arow][acol]);
    stage16(ga + k0 + 8, &sA[arow][acol + 8]);
    stage16(gb + k0,     &sB[brow][bcol]);
    if (k0 + BK < K) {  // gfx1250 global_prefetch_b8 of next K-tile
      __builtin_prefetch((const void*)(ga + k0 + BK), 0, 1);
      __builtin_prefetch((const void*)(gb + k0 + BK), 0, 1);
    }
    stage_wait();
    __syncthreads();

    const v16bf af0 = load_a_frag(&sA[wm * 32 +  0][0], BK + PAD, lane);
    const v16bf af1 = load_a_frag(&sA[wm * 32 + 16][0], BK + PAD, lane);
    const v16bf bf0 = load_b_frag(&sB[wn * 32 +  0][0], BK + PAD, lane);
    const v16bf bf1 = load_b_frag(&sB[wn * 32 + 16][0], BK + PAD, lane);
    acc[0][0] = WMMA_BF16(af0, bf0, acc[0][0]);
    acc[0][1] = WMMA_BF16(af0, bf1, acc[0][1]);
    acc[1][0] = WMMA_BF16(af1, bf0, acc[1][0]);
    acc[1][1] = WMMA_BF16(af1, bf1, acc[1][1]);
    __syncthreads();
  }

  // C/D layout: lane holds col N=lane&15; row = (lane>>4)*8 + r for VGPR r.
  const int nloc  = lane & 15;
  const int half8 = (lane >> 4) << 3;
#pragma unroll
  for (int i = 0; i < 2; ++i)
#pragma unroll
    for (int j = 0; j < 2; ++j) {
      const size_t col = (size_t)(n0 + wn * 32 + j * 16 + nloc);
#pragma unroll
      for (int r = 0; r < 8; ++r) {
        const size_t row = (size_t)(m0 + wm * 32 + i * 16 + half8 + r);
        if (OUT_F32) ((float*)Out)[row * N + col] = acc[i][j][r];
        else         ((__bf16*)Out)[row * N + col] = (__bf16)acc[i][j][r];
      }
    }
}

// ---------------------------------------------------------------------------
// Flash attention (causal, GQA). One block = 64 query rows of one (b,h).
// 4 waves x 16 query rows. Key loop in tiles of 32. All matmuls via WMMA.
// ---------------------------------------------------------------------------
__global__ __launch_bounds__(128) void attn_fwd(const __bf16* __restrict__ Q,
                                                const __bf16* __restrict__ Kv,
                                                const __bf16* __restrict__ V,
                                                __bf16* __restrict__ O) {
  constexpr int PAD = 8;
  __shared__ __bf16 sQ[64][HD + PAD];        // query tile, row-major
  __shared__ __bf16 sK[32][HD + PAD];        // key tile, row-major (B-frag ready)
  __shared__ __bf16 sVt[HD][32 + PAD];       // value tile TRANSPOSED [d][key]
  __shared__ __bf16 sP[4][16][32 + PAD];     // per-wave P (D-layout -> A-layout)

  const int tid  = threadIdx.x;
  const int lane = tid & 31;
  const int w    = tid >> 5;                 // wave id 0..3 -> q sub-tile
  const int qblk = blockIdx.x;               // 0..SEQ/64-1
  const int bh   = blockIdx.y;               // 0..B*NH-1
  const int b    = bh >> 4;
  const int h    = bh & 15;
  const int kvh  = h >> 2;                   // GQA: 4 query heads per kv head

  const size_t qrow0 = (size_t)b * SEQ + (size_t)qblk * 64;

  // Stage Q tile (64 x 128 bf16) via async-to-LDS when available
  for (int c = tid; c < 64 * 16; c += 128) {
    const int row = c >> 4, col = (c & 15) << 3;
    stage16(Q + (qrow0 + row) * EMB + h * HD + col, &sQ[row][col]);
  }
  stage_wait();
  __syncthreads();

  // Preload Q A-fragments (4 chunks of K=32 over head dim)
  v16bf qf[4];
#pragma unroll
  for (int dc = 0; dc < 4; ++dc) qf[dc] = load_a_frag(&sQ[w * 16][dc * 32], HD + PAD, lane);

  const v8f vzero = {0.f, 0.f, 0.f, 0.f, 0.f, 0.f, 0.f, 0.f};
  v8f oacc[8];
  float mrow[8], lrow[8];
#pragma unroll
  for (int d = 0; d < 8; ++d) oacc[d] = vzero;
#pragma unroll
  for (int r = 0; r < 8; ++r) { mrow[r] = -1e30f; lrow[r] = 0.f; }

  const int nloc   = lane & 15;
  const int half8  = (lane >> 4) << 3;
  const int qrow_l = qblk * 64 + w * 16 + half8;  // global q row of element r=0
  const int nkb    = (qblk + 1) * 2;              // causal: keys up to end of q tile

  for (int kb = 0; kb < nkb; ++kb) {
    const size_t krow0 = (size_t)b * SEQ + (size_t)kb * 32;
    // Stage K (async to LDS, row-major) and V (register transpose): 32x128 each
    for (int c = tid; c < 32 * 16; c += 128) {
      const int row = c >> 4, col = (c & 15) << 3;
      stage16(Kv + (krow0 + row) * KVE + kvh * HD + col, &sK[row][col]);
      const v8bf vv = *(const v8bf*)(V + (krow0 + row) * KVE + kvh * HD + col);
#pragma unroll
      for (int i = 0; i < 8; ++i) sVt[col + i][row] = vv[i];
    }
    stage_wait();
    __syncthreads();

    // S = Q @ K^T : preload all 8 K B-fragments into independent registers
    // so the WMMA chain isn't serialized on one ds destination group.
    v16bf kf[8];
#pragma unroll
    for (int dc = 0; dc < 4; ++dc) {
      kf[dc]     = load_b_frag(&sK[ 0][dc * 32], HD + PAD, lane);
      kf[dc + 4] = load_b_frag(&sK[16][dc * 32], HD + PAD, lane);
    }
    v8f s0 = vzero, s1 = vzero;
#pragma unroll
    for (int dc = 0; dc < 4; ++dc) {
      s0 = WMMA_BF16(qf[dc], kf[dc],     s0);
      s1 = WMMA_BF16(qf[dc], kf[dc + 4], s1);
    }

    // Online softmax (per-row stats; rows split across lane halves, so the
    // row reduction is a shfl_xor over masks 1,2,4,8 within each 16-lane half)
#pragma unroll
    for (int r = 0; r < 8; ++r) {
      const int qg = qrow_l + r;
      float a0 = s0[r] * SCALE;
      float a1 = s1[r] * SCALE;
      if (kb * 32 + nloc      > qg) a0 = -1e30f;  // causal mask
      if (kb * 32 + 16 + nloc > qg) a1 = -1e30f;
      float rm = fmaxf(a0, a1);
#pragma unroll
      for (int off = 1; off < 16; off <<= 1) rm = fmaxf(rm, __shfl_xor(rm, off, 32));
      const float mn    = fmaxf(mrow[r], rm);
      const float alpha = __expf(mrow[r] - mn);
      a0 = __expf(a0 - mn);
      a1 = __expf(a1 - mn);
      float rs = a0 + a1;
#pragma unroll
      for (int off = 1; off < 16; off <<= 1) rs += __shfl_xor(rs, off, 32);
      lrow[r] = lrow[r] * alpha + rs;
      mrow[r] = mn;
#pragma unroll
      for (int d = 0; d < 8; ++d) oacc[d][r] *= alpha;
      // D-layout -> row-major LDS (A-layout source) for the PV matmul
      sP[w][half8 + r][nloc]      = (__bf16)a0;
      sP[w][half8 + r][16 + nloc] = (__bf16)a1;
    }
    __syncthreads();

    // O += P @ V : A = P (16q x 32k), B = V^T-staged (32k x 16d per tile)
    const v16bf pf = load_a_frag(&sP[w][0][0], 32 + PAD, lane);
    v16bf vf[8];
#pragma unroll
    for (int d = 0; d < 8; ++d) vf[d] = load_b_frag(&sVt[d * 16][0], 32 + PAD, lane);
#pragma unroll
    for (int d = 0; d < 8; ++d) oacc[d] = WMMA_BF16(pf, vf[d], oacc[d]);
    __syncthreads();  // protect sK/sVt before next tile's loads
  }

  // Normalize and write O (bf16, packed back into (B*T, EMB) head-major)
#pragma unroll
  for (int r = 0; r < 8; ++r) {
    const float inv  = 1.f / lrow[r];
    const size_t row = qrow0 + (size_t)(w * 16 + half8 + r);
#pragma unroll
    for (int d = 0; d < 8; ++d) {
      O[row * EMB + h * HD + d * 16 + nloc] = (__bf16)(oacc[d][r] * inv);
    }
  }
}

// ---------------------------------------------------------------------------
// Launch
// ---------------------------------------------------------------------------
extern "C" void kernel_launch(void* const* d_in, const int* in_sizes, int n_in,
                              void* d_out, int out_size, void* d_ws, size_t ws_size,
                              hipStream_t stream) {
  (void)in_sizes; (void)n_in; (void)out_size; (void)ws_size;
  const float* x  = (const float*)d_in[0];
  const float* wq = (const float*)d_in[1];
  const float* wk = (const float*)d_in[2];
  const float* wv = (const float*)d_in[3];
  const float* wo = (const float*)d_in[4];
  float* out = (float*)d_out;

  char* ws = (char*)d_ws;
  size_t off = 0;
  auto bump = [&](size_t elems) -> __bf16* {
    void* p = (void*)(ws + off);
    off += ((elems * 2) + 255) & ~(size_t)255;
    return (__bf16*)p;
  };
  __bf16* xb  = bump((size_t)MTOT * EMB);
  __bf16* wqb = bump((size_t)EMB * EMB);
  __bf16* wkb = bump((size_t)KVE * EMB);
  __bf16* wvb = bump((size_t)KVE * EMB);
  __bf16* wob = bump((size_t)EMB * EMB);
  __bf16* qb  = bump((size_t)MTOT * EMB);
  __bf16* kb_ = bump((size_t)MTOT * KVE);
  __bf16* vb  = bump((size_t)MTOT * KVE);
  __bf16* ob  = bump((size_t)MTOT * EMB);

  auto cvt = [&](const float* src, __bf16* dst, size_t n) {
    const int n4 = (int)(n / 4);
    cvt_f32_bf16<<<dim3((n4 + 255) / 256), dim3(256), 0, stream>>>(src, dst, n4);
  };
  cvt(x,  xb,  (size_t)MTOT * EMB);
  cvt(wq, wqb, (size_t)EMB * EMB);
  cvt(wk, wkb, (size_t)KVE * EMB);
  cvt(wv, wvb, (size_t)KVE * EMB);
  cvt(wo, wob, (size_t)EMB * EMB);

  // Projections: q = x @ wq^T, k = x @ wk^T, v = x @ wv^T   (bf16 out)
  gemm_bf16<false><<<dim3(EMB / 64, MTOT / 128), 256, 0, stream>>>(xb, wqb, (void*)qb, MTOT, EMB, EMB);
  gemm_bf16<false><<<dim3(KVE / 64, MTOT / 128), 256, 0, stream>>>(xb, wkb, (void*)kb_, MTOT, KVE, EMB);
  gemm_bf16<false><<<dim3(KVE / 64, MTOT / 128), 256, 0, stream>>>(xb, wvb, (void*)vb, MTOT, KVE, EMB);

  // Causal GQA flash attention
  attn_fwd<<<dim3(SEQ / 64, BATCH * NH), 128, 0, stream>>>(qb, kb_, vb, ob);

  // Output projection: y = o @ wo^T   (fp32 out)
  gemm_bf16<true><<<dim3(EMB / 64, MTOT / 128), 256, 0, stream>>>(ob, wob, (void*)out, MTOT, EMB, EMB);
}
